// TDID_6571299963632
// MI455X (gfx1250) — matrix operand order repaired and checked by reading, not compiled, not run on hardware
//
#include <hip/hip_runtime.h>
#include <math.h>

// ---------------------------------------------------------------------------
// Types
// ---------------------------------------------------------------------------
typedef _Float16 half_t;
typedef __attribute__((ext_vector_type(16))) _Float16 v16h;
typedef __attribute__((ext_vector_type(8)))  _Float16 v8h;
typedef __attribute__((ext_vector_type(8)))  float    v8f;

#define CDIV(a, b) ((unsigned)(((size_t)(a) + (size_t)(b) - 1) / (size_t)(b)))

// ---------------------------------------------------------------------------
// 3x3 conv as implicit GEMM via V_WMMA_F32_16X16X32_F16.
//   in    : NHWC fp16, channels = Cpad (multiple of 32)
//   wpack : [Cout/16][3][3][Cpad/32] tiles of 512 halves in WMMA B lane layout
//   zpage : Cpad halves of zeros (halo reads redirect here -> no divergence)
//   out   : NHWC fp16, channels = Cout (multiple of 64)
// One wave per block; wave computes a 16(M pixels) x 64(N channels) tile:
// four f32 accumulators share one A fragment, giving 4 independent WMMA
// chains per iteration and no EXEC manipulation anywhere near the WMMAs.
// ---------------------------------------------------------------------------
__global__ __launch_bounds__(32)
void conv3x3_wmma(const half_t* __restrict__ in, const half_t* __restrict__ wpack,
                  const float* __restrict__ bias, half_t* __restrict__ out,
                  const half_t* __restrict__ zpage,
                  int N, int H, int W, int Cpad, int Cout, int relu)
{
    const int lane = threadIdx.x;
    const int kgrp = lane >> 4;       // 0: K 0..15, 1: K 16..31
    const int rowM = lane & 15;
    const int total = N * H * W;
    const int m0  = blockIdx.x * 16;
    const int nt0 = blockIdx.y * 4;   // four consecutive 16-wide N tiles
    const int KC  = Cpad >> 5;        // number of K=32 chunks per tap

    int p = m0 + rowM;
    const bool prow_ok = (p < total);
    int pp = prow_ok ? p : 0;
    int x = pp % W;
    int t = pp / W;
    int y = t % H;
    int n = t / H;

    v8f acc[4];
    #pragma unroll
    for (int j = 0; j < 4; ++j) acc[j] = (v8f){};

    const size_t ldw = (size_t)9 * KC * 512;                    // halves per N tile
    const half_t* wb[4];
    #pragma unroll
    for (int j = 0; j < 4; ++j)
        wb[j] = wpack + (size_t)(nt0 + j) * ldw + (size_t)lane * 16;

    const int aoff0 = kgrp * 16;
    const int aoff1 = kgrp * 16 + 8;

    for (int ky = 0; ky < 3; ++ky) {
        const int iy = y + ky - 1;
        const bool oy = (unsigned)iy < (unsigned)H;
        for (int kx = 0; kx < 3; ++kx) {
            const int ix = x + kx - 1;
            const bool ok = prow_ok && oy && ((unsigned)ix < (unsigned)W);
            // Halo lanes read a zero page: loads stay unconditional, EXEC
            // stays all-ones, and no fragment register is ever zero-filled
            // by VALU (avoids WMMA data hazard NOPs).
            const half_t* abase =
                ok ? (in + ((size_t)(n * H + iy) * W + ix) * Cpad) : zpage;
            const size_t wk = (size_t)(ky * 3 + kx) * KC * 512;
            for (int kc = 0; kc < KC; ++kc) {
                union { v16h v; v8h h[2]; } a;
                const half_t* s = abase + kc * 32;
                a.h[0] = *(const v8h*)(s + aoff0);
                a.h[1] = *(const v8h*)(s + aoff1);
                __builtin_prefetch(s + 32, 0, 0);   // speculative, next K chunk
                #pragma unroll
                for (int j = 0; j < 4; ++j) {
                    v16h b = *(const v16h*)(wb[j] + wk + (size_t)kc * 512);
                    acc[j] = __builtin_amdgcn_wmma_f32_16x16x32_f16(
                        false, a.v, false, b, (short)0, acc[j], false, false);
                }
            }
        }
    }

    // C/D layout: lane holds column N = lane&15; VGPR r holds row M = r + 8*(lane>>4)
    float bs[4];
    int   cn[4];
    #pragma unroll
    for (int j = 0; j < 4; ++j) {
        cn[j] = (nt0 + j) * 16 + (lane & 15);
        bs[j] = bias[cn[j]];
    }
    for (int r = 0; r < 8; ++r) {
        const int pr = m0 + r + kgrp * 8;
        if (pr < total) {
            half_t* o = out + (size_t)pr * Cout;
            #pragma unroll
            for (int j = 0; j < 4; ++j) {
                float v = acc[j][r] + bs[j];
                if (relu) v = fmaxf(v, 0.f);
                o[cn[j]] = (half_t)v;
            }
        }
    }
}

// ---------------------------------------------------------------------------
// Pack OIHW fp32 weights into the WMMA B-fragment layout (fp16), zero-padding
// input channels up to Cpad. Element e of lane l in tile (nt, ky, kx, kc):
//   outch = nt*16 + (l&15),  inch = kc*32 + (l>>4)*16 + e
// ---------------------------------------------------------------------------
__global__ void pack_w(const float* __restrict__ w, half_t* __restrict__ out,
                       int Cout, int Cin, int Cpad)
{
    const int KC = Cpad >> 5;
    const size_t total = (size_t)Cout * 9 * Cpad;
    size_t idx = (size_t)blockIdx.x * blockDim.x + threadIdx.x;
    if (idx >= total) return;
    const int e    = (int)(idx & 15);
    const int lane = (int)((idx >> 4) & 31);
    size_t t = idx >> 9;
    const int kc = (int)(t % KC); t /= KC;
    const int kk = (int)(t % 9);  t /= 9;
    const int nt = (int)t;
    const int ky = kk / 3, kx = kk % 3;
    const int oc = nt * 16 + (lane & 15);
    const int ic = kc * 32 + (lane >> 4) * 16 + e;
    float v = 0.f;
    if (ic < Cin) v = w[(((size_t)oc * Cin + ic) * 3 + ky) * 3 + kx];
    out[idx] = (half_t)v;
}

// ---------------------------------------------------------------------------
// Convert NHWC fp32 (Cin channels) -> NHWC fp16 padded to Cpad channels.
// ---------------------------------------------------------------------------
__global__ void cvt_pad(const float* __restrict__ in, half_t* __restrict__ out,
                        int NHW, int Cin, int Cpad)
{
    size_t idx = (size_t)blockIdx.x * blockDim.x + threadIdx.x;
    const size_t total = (size_t)NHW * Cpad;
    if (idx >= total) return;
    const int c = (int)(idx % Cpad);
    const size_t p = idx / Cpad;
    out[idx] = (c < Cin) ? (half_t)in[p * Cin + c] : (half_t)0;
}

// ---------------------------------------------------------------------------
// 2x2/2 maxpool, NHWC fp16.
// ---------------------------------------------------------------------------
__global__ void maxpool2(const half_t* __restrict__ in, half_t* __restrict__ out,
                         int N, int H, int W, int C)
{
    const int Ho = H >> 1, Wo = W >> 1;
    const size_t total = (size_t)N * Ho * Wo * C;
    size_t idx = (size_t)blockIdx.x * blockDim.x + threadIdx.x;
    if (idx >= total) return;
    const int c = (int)(idx % C);
    size_t t = idx / C;
    const int x = (int)(t % Wo); t /= Wo;
    const int y = (int)(t % Ho);
    const int n = (int)(t / Ho);
    const half_t* p = in + (((size_t)(n * H + 2 * y) * W) + 2 * x) * C + c;
    const size_t rs = (size_t)W * C;
    float a = (float)p[0], b = (float)p[C], cc = (float)p[rs], d = (float)p[rs + C];
    out[idx] = (half_t)fmaxf(fmaxf(a, b), fmaxf(cc, d));
}

// ---------------------------------------------------------------------------
// Grouped cross-correlation (reference semantics of feature_group_count=512
// with 1024 filters): output channel o uses img channel o>>1 and target
// filter o (= tgt[o / C][.][.][o % C]). Output NHWC fp16 [H,W,2C].
// ---------------------------------------------------------------------------
__global__ void xcorr_kernel(const half_t* __restrict__ img, const half_t* __restrict__ tgt,
                             half_t* __restrict__ out, int H, int W, int C, int kh, int kw)
{
    const size_t total = (size_t)H * W * 2 * C;
    size_t idx = (size_t)blockIdx.x * blockDim.x + threadIdx.x;
    if (idx >= total) return;
    const int o = (int)(idx % (2 * C));
    size_t t = idx / (2 * C);
    const int x = (int)(t % W);
    const int y = (int)(t / W);
    const int imgc = o >> 1;
    const int tt = o / C;
    const int cc = o % C;
    float acc = 0.f;
    for (int dy = 0; dy < kh; ++dy) {
        const int iy = y + dy - kh / 2;
        if ((unsigned)iy >= (unsigned)H) continue;
        for (int dx = 0; dx < kw; ++dx) {
            const int ix = x + dx - kw / 2;
            if ((unsigned)ix >= (unsigned)W) continue;
            acc += (float)img[((size_t)(iy * W + ix)) * C + imgc] *
                   (float)tgt[(((size_t)tt * kh + dy) * kw + dx) * C + cc];
        }
    }
    out[idx] = (half_t)acc;
}

// ---------------------------------------------------------------------------
// 1x1 conv head: NHWC fp16 input -> NCHW fp32 output [Co][HW].
// ---------------------------------------------------------------------------
__global__ void head1x1(const half_t* __restrict__ in, const float* __restrict__ w,
                        const float* __restrict__ b, float* __restrict__ out,
                        int HW, int Cin, int Co)
{
    const int idx = blockIdx.x * blockDim.x + threadIdx.x;
    if (idx >= Co * HW) return;
    const int p  = idx % HW;
    const int co = idx / HW;
    const half_t* ip = in + (size_t)p * Cin;
    const float*  wp = w + (size_t)co * Cin;
    float acc = b[co];
    for (int c = 0; c < Cin; ++c) acc += (float)ip[c] * wp[c];
    out[idx] = acc;
}

// ---------------------------------------------------------------------------
// Paired softmax over channels (c, c+half_c), NCHW fp32 [2*half_c][HW].
// ---------------------------------------------------------------------------
__global__ void softmax_pairs(const float* __restrict__ cls, float* __restrict__ prob,
                              int HW, int half_c)
{
    const int idx = blockIdx.x * blockDim.x + threadIdx.x;
    if (idx >= half_c * HW) return;
    const int p = idx % HW;
    const int c = idx / HW;
    const float a = cls[(size_t)c * HW + p];
    const float b = cls[(size_t)(c + half_c) * HW + p];
    const float m = fmaxf(a, b);
    const float ea = __expf(a - m), eb = __expf(b - m);
    const float inv = 1.f / (ea + eb);
    prob[(size_t)c * HW + p] = ea * inv;
    prob[(size_t)(c + half_c) * HW + p] = eb * inv;
}

// ---------------------------------------------------------------------------
// Host orchestration.
// Input order (setup_inputs dict, tuples flattened):
//  0: target_data (2,80,80,3)   1: im_data (1,384,384,3)
//  2..14: vgg_w[0..12]          15..27: vgg_b[0..12]
//  28: conv1_w (512,1024,3,3)   29: conv1_b (512)
//  30: score_w (18,512,1,1)     31: score_b (18)
//  32: bbox_w  (36,512,1,1)     33: bbox_b  (36)
// ---------------------------------------------------------------------------
extern "C" void kernel_launch(void* const* d_in, const int* in_sizes, int n_in,
                              void* d_out, int out_size, void* d_ws, size_t ws_size,
                              hipStream_t stream)
{
    (void)in_sizes; (void)n_in; (void)out_size; (void)ws_size;

    const float* target  = (const float*)d_in[0];
    const float* im      = (const float*)d_in[1];
    const float* vggW[13];
    const float* vggB[13];
    for (int i = 0; i < 13; ++i) {
        vggW[i] = (const float*)d_in[2 + i];
        vggB[i] = (const float*)d_in[15 + i];
    }
    const float* conv1_w = (const float*)d_in[28];
    const float* conv1_b = (const float*)d_in[29];
    const float* score_w = (const float*)d_in[30];
    const float* score_b = (const float*)d_in[31];
    const float* bbox_w  = (const float*)d_in[32];
    const float* bbox_b  = (const float*)d_in[33];
    float* out = (float*)d_out;

    static const int CIN[13]  = {3, 64, 64, 128, 128, 256, 256, 256, 512, 512, 512, 512, 512};
    static const int COUT[13] = {64, 64, 128, 128, 256, 256, 256, 512, 512, 512, 512, 512, 512};
    static const int CPAD[13] = {32, 64, 64, 128, 128, 256, 256, 256, 512, 512, 512, 512, 512};

    // ---- workspace layout -------------------------------------------------
    char* ws = (char*)d_ws;
    size_t off = 0;
    auto alloc = [&](size_t bytes) -> void* {
        void* p = ws + off;
        off += (bytes + 255) & ~(size_t)255;
        return p;
    };

    half_t* wp[13];
    for (int i = 0; i < 13; ++i)
        wp[i] = (half_t*)alloc((size_t)COUT[i] * 9 * CPAD[i] * sizeof(half_t));
    half_t* wp_rpn  = (half_t*)alloc((size_t)512 * 9 * 1024 * sizeof(half_t));
    const size_t BUF_HALVES = (size_t)384 * 384 * 64;       // largest activation
    half_t* bufA    = (half_t*)alloc(BUF_HALVES * sizeof(half_t));
    half_t* bufB    = (half_t*)alloc(BUF_HALVES * sizeof(half_t));
    half_t* tgtFeat = (half_t*)alloc((size_t)2 * 5 * 5 * 512 * sizeof(half_t));
    half_t* ccBuf   = (half_t*)alloc((size_t)24 * 24 * 1024 * sizeof(half_t));
    half_t* rpnBuf  = (half_t*)alloc((size_t)24 * 24 * 512 * sizeof(half_t));
    float*  clsBuf  = (float*)alloc((size_t)18 * 576 * sizeof(float));
    half_t* zpage   = (half_t*)alloc((size_t)1024 * sizeof(half_t));   // halo zeros

    hipMemsetAsync(zpage, 0, 1024 * sizeof(half_t), stream);

    // ---- pack all conv weights into WMMA fragment layout ------------------
    for (int i = 0; i < 13; ++i) {
        const size_t tot = (size_t)COUT[i] * 9 * CPAD[i];
        pack_w<<<CDIV(tot, 256), 256, 0, stream>>>(vggW[i], wp[i], COUT[i], CIN[i], CPAD[i]);
    }
    {
        const size_t tot = (size_t)512 * 9 * 1024;
        pack_w<<<CDIV(tot, 256), 256, 0, stream>>>(conv1_w, wp_rpn, 512, 1024, 1024);
    }

    // ---- VGG trunk (shared weights, run on targets then image) ------------
    auto run_vgg = [&](const float* src, int N, int H0, int W0) -> half_t* {
        int h = H0, w = W0;
        const size_t nhw = (size_t)N * h * w;
        cvt_pad<<<CDIV(nhw * 32, 256), 256, 0, stream>>>(src, bufA, (int)nhw, 3, 32);
        half_t* cur = bufA;
        half_t* nxt = bufB;
        for (int i = 0; i < 13; ++i) {
            const size_t tot = (size_t)N * h * w;
            dim3 g(CDIV(tot, 16), (unsigned)(COUT[i] / 64));
            conv3x3_wmma<<<g, 32, 0, stream>>>(cur, wp[i], vggB[i], nxt, zpage,
                                               N, h, w, CPAD[i], COUT[i], 1);
            half_t* tmp = cur; cur = nxt; nxt = tmp;
            if (i == 1 || i == 3 || i == 6 || i == 9) {
                const size_t pt = (size_t)N * (h / 2) * (w / 2) * COUT[i];
                maxpool2<<<CDIV(pt, 256), 256, 0, stream>>>(cur, nxt, N, h, w, COUT[i]);
                tmp = cur; cur = nxt; nxt = tmp;
                h >>= 1; w >>= 1;
            }
        }
        return cur;
    };

    half_t* tf = run_vgg(target, 2, 80, 80);                   // -> [2,5,5,512]
    hipMemcpyAsync(tgtFeat, tf, (size_t)2 * 5 * 5 * 512 * sizeof(half_t),
                   hipMemcpyDeviceToDevice, stream);
    half_t* imf = run_vgg(im, 1, 384, 384);                    // -> [1,24,24,512]

    // ---- cross-correlation -> RPN conv -> heads ---------------------------
    {
        const size_t tot = (size_t)24 * 24 * 1024;
        xcorr_kernel<<<CDIV(tot, 256), 256, 0, stream>>>(imf, tgtFeat, ccBuf,
                                                         24, 24, 512, 5, 5);
    }
    {
        dim3 g(CDIV((size_t)24 * 24, 16), 512 / 64);
        conv3x3_wmma<<<g, 32, 0, stream>>>(ccBuf, wp_rpn, conv1_b, rpnBuf, zpage,
                                           1, 24, 24, 1024, 512, 0 /* no relu */);
    }
    head1x1<<<CDIV(18 * 576, 256), 256, 0, stream>>>(rpnBuf, score_w, score_b,
                                                     clsBuf, 576, 512, 18);
    head1x1<<<CDIV(36 * 576, 256), 256, 0, stream>>>(rpnBuf, bbox_w, bbox_b,
                                                     out + 18 * 576, 576, 512, 36);
    softmax_pairs<<<CDIV(9 * 576, 256), 256, 0, stream>>>(clsBuf, out, 576, 9);
}
